// FourierForwardNetwork_71305047048412
// MI455X (gfx1250) — matrix operand order repaired and checked by reading, not compiled, not run on hardware
//
#include <hip/hip_runtime.h>
#include <math.h>

typedef __attribute__((ext_vector_type(16))) _Float16 v16h;
typedef __attribute__((ext_vector_type(2)))  _Float16 v2h;
typedef __attribute__((ext_vector_type(8)))  float    v8f;
typedef __attribute__((ext_vector_type(4)))  float    v4f;
typedef __attribute__((ext_vector_type(4)))  unsigned int v4u;

#define DDIM   128
#define TWO_PI 6.28318530717958647692f

// ---------------------------------------------------------------------------
// Kernel 1: c[d] = (1/128) * sum_k ( br[k]*cos(2*pi*k*d/128) - bi[k]*sin(...) )
// Circulant kernel of ifft(fft(x)*b).real.   One block, 128 threads.
// ---------------------------------------------------------------------------
__global__ void fourier_build_c(const float* __restrict__ br,
                                const float* __restrict__ bi,
                                float* __restrict__ c) {
  int d = threadIdx.x;
  float acc = 0.0f;
  for (int k = 0; k < DDIM; ++k) {
    float ang = (TWO_PI / (float)DDIM) * (float)((k * d) & (DDIM - 1));
    acc += br[k] * __cosf(ang) - bi[k] * __sinf(ang);
  }
  c[d] = acc * (1.0f / (float)DDIM);
}

// ---------------------------------------------------------------------------
// Kernel 2: out[r][n] = sum_k x[r][k] * Bmat[k][n],  Bmat[k][n] = c[(n-k)&127]
// WMMA f32_16x16x32_f16.  8 waves/WG.  B stored fragment-major in LDS so each
// B fragment is two contiguous ds_load_b128 per lane.  Two 16-row tiles per
// wave iteration share every B fragment.  Tail tile handled branchlessly by
// clamping (duplicate identical stores — deterministic).
// ---------------------------------------------------------------------------
__global__ __launch_bounds__(256, 4) void fourier_gemm(
    const float* __restrict__ x, const float* __restrict__ c,
    float* __restrict__ out, int numTiles, int totalWaves) {
  // Bf[frag][lane][word]: frag = nt*4+kb (32 frags), 8 words = 16 f16 halves
  // per lane = the exact per-lane B-fragment register image (32 B, contiguous).
  __shared__ __align__(16) unsigned int Bf[32 * 32 * 8];

  const int t = threadIdx.x;
  for (int i = t; i < 32 * 32 * 8; i += 256) {
    const int frag = i >> 8;          // 0..31
    const int ln   = (i >> 3) & 31;   // lane this word belongs to
    const int j    = i & 7;           // word within lane fragment
    const int nt   = frag >> 2;
    const int kb   = frag & 3;
    const int n    = nt * 16 + (ln & 15);
    const int k0   = kb * 32 + (ln >> 4) * 8 + 2 * j + ((j >= 4) ? 8 : 0);
    v2h pk;
    pk[0] = (_Float16)c[(n - k0)     & (DDIM - 1)];
    pk[1] = (_Float16)c[(n - k0 - 1) & (DDIM - 1)];
    Bf[i] = __builtin_bit_cast(unsigned int, pk);
  }
  __syncthreads();

  const int lane = t & 31;
  const int wave = blockIdx.x * 8 + (t >> 5);
  const int m    = lane & 15;   // A row within tile / C column within N-tile
  const int hi16 = lane >> 4;   // half-wave select
  const int off8 = hi16 * 8;    // A-fragment K offset for upper half-wave

  for (int tile0 = wave * 2; tile0 < numTiles; tile0 += totalWaves * 2) {
    // Branchless tail: clamp. Odd-count last wave computes acc1==acc0 and
    // stores the same values to the same addresses (same lane, in order).
    const int tile1 = (tile0 + 1 < numTiles) ? (tile0 + 1) : (numTiles - 1);

    // ---- load two 16x128 A panels straight into f16 A-fragments ----
    v16h a[2][4];
    const int tiles[2] = {tile0, tile1};
#pragma unroll
    for (int tt = 0; tt < 2; ++tt) {
      const float* xr = x + (size_t)(tiles[tt] * 16 + m) * DDIM;
#pragma unroll
      for (int kb = 0; kb < 4; ++kb) {
        const float* base = xr + kb * 32;
        v4f s0 = __builtin_nontemporal_load((const v4f*)(base + off8));
        v4f s1 = __builtin_nontemporal_load((const v4f*)(base + off8 + 4));
        v4f s2 = __builtin_nontemporal_load((const v4f*)(base + 16 + off8));
        v4f s3 = __builtin_nontemporal_load((const v4f*)(base + 16 + off8 + 4));
#pragma unroll
        for (int j = 0; j < 4; ++j) {
          a[tt][kb][j]      = (_Float16)s0[j];   // K = off8 + j
          a[tt][kb][4 + j]  = (_Float16)s1[j];   // K = off8 + 4 + j
          a[tt][kb][8 + j]  = (_Float16)s2[j];   // K = 16 + off8 + j
          a[tt][kb][12 + j] = (_Float16)s3[j];   // K = 16 + off8 + 4 + j
        }
      }
    }

    float* orow0 = out + (size_t)tile0 * 16 * DDIM;
    float* orow1 = out + (size_t)tile1 * 16 * DDIM;
#pragma unroll
    for (int nt = 0; nt < 8; ++nt) {
      v8f acc0 = {};
      v8f acc1 = {};
#pragma unroll
      for (int kb = 0; kb < 4; ++kb) {
        // One B fragment = 32 contiguous bytes per lane: two ds_load_b128.
        const unsigned int* bp = &Bf[(((nt << 2) | kb) * 32 + lane) * 8];
        union { v4u q[2]; v16h h; } bu;
        bu.q[0] = *(const v4u*)(bp);
        bu.q[1] = *(const v4u*)(bp + 4);
        const v16h b = bu.h;
        acc0 = __builtin_amdgcn_wmma_f32_16x16x32_f16(
            false, a[0][kb], false, b, (short)0, acc0, false, false);
        acc1 = __builtin_amdgcn_wmma_f32_16x16x32_f16(
            false, a[1][kb], false, b, (short)0, acc1, false, false);
      }
      const int col = nt * 16 + m;
#pragma unroll
      for (int v = 0; v < 8; ++v) {
        const size_t roff = (size_t)(hi16 * 8 + v) * DDIM + col;
        __builtin_nontemporal_store(acc0[v], &orow0[roff]);
        __builtin_nontemporal_store(acc1[v], &orow1[roff]);
      }
    }
  }
}

// ---------------------------------------------------------------------------
// Launch
// ---------------------------------------------------------------------------
extern "C" void kernel_launch(void* const* d_in, const int* in_sizes, int n_in,
                              void* d_out, int out_size, void* d_ws, size_t ws_size,
                              hipStream_t stream) {
  const float* x  = (const float*)d_in[0];   // [B,S,D] f32
  const float* br = (const float*)d_in[3];   // b_real [D]
  const float* bi = (const float*)d_in[4];   // b_imag [D]
  float* out = (float*)d_out;
  float* c   = (float*)d_ws;                 // 128 f32 scratch

  const int rows     = in_sizes[0] / DDIM;   // 262144
  const int numTiles = rows / 16;            // 16384

  fourier_build_c<<<1, DDIM, 0, stream>>>(br, bi, c);

  int blocks = 1024;                         // 8192 waves -> 2 tiles per wave
  const int pairs = (numTiles + 1) / 2;
  if (blocks * 8 > pairs) blocks = (pairs + 7) / 8;
  const int totalWaves = blocks * 8;
  fourier_gemm<<<blocks, 256, 0, stream>>>(x, c, out, numTiles, totalWaves);
}
